// SpinorGate_58265526337698
// MI455X (gfx1250) — matrix operand order repaired and checked by reading, not compiled, not run on hardware
//
#include <hip/hip_runtime.h>
#include <hip/hip_bf16.h>

typedef __attribute__((ext_vector_type(16))) _Float16 v16h;
typedef __attribute__((ext_vector_type(8)))  _Float16 v8h;
typedef __attribute__((ext_vector_type(8)))  float    v8f;
typedef __attribute__((ext_vector_type(4)))  float    v4f;

#define GATE_SCALE 1.7015f
// fixed problem dims from the reference
#define BS   2048                 // B*S
#define DD   1024                 // D
#define K2   2048                 // 2*D (GEMM K)
#define NB   128                  // number of 8-blocks
#define BSD  (2u * 1024u * 1024u) // B*S*D
#define NEUMANN_TERMS 6           // ||X|| <~ 0.12 -> err ~ 2*||X||^7 ~ 1e-7

// ---------------- staging: f32 -> f16 copies of x_cat and W ----------------
__global__ void cvt_x_kernel(const float* __restrict__ xr, const float* __restrict__ xi,
                             _Float16* __restrict__ xh) {
  int idx = blockIdx.x * blockDim.x + threadIdx.x;  // BS*K2 elements
  int m = idx >> 11, k = idx & (K2 - 1);
  float v = (k < DD) ? xr[m * DD + k] : xi[m * DD + (k - DD)];
  xh[idx] = (_Float16)v;
}

__global__ void cvt_w_kernel(const float* __restrict__ W, _Float16* __restrict__ Wh) {
  int idx = blockIdx.x * blockDim.x + threadIdx.x;  // DD*K2 elements
  Wh[idx] = (_Float16)W[idx];
}

// ---------------- fragment loaders ----------------
__device__ inline v16h load_frag_f16(const _Float16* p0, const _Float16* p1) {
  // p0 -> elements 0..7, p1 -> elements 8..15 (each a 16B contiguous run)
  v8h a0 = *(const v8h*)p0;
  v8h a1 = *(const v8h*)p1;
  v16h r;
#pragma unroll
  for (int e = 0; e < 8; ++e) { r[e] = a0[e]; r[8 + e] = a1[e]; }
  return r;
}

__device__ inline v16h load_frag_f32(const float* p0, const float* p1) {
  v4f a0 = *(const v4f*)p0; v4f a1 = *(const v4f*)(p0 + 4);
  v4f b0 = *(const v4f*)p1; v4f b1 = *(const v4f*)(p1 + 4);
  v16h r;
#pragma unroll
  for (int e = 0; e < 4; ++e) {
    r[e]      = (_Float16)a0[e];
    r[4 + e]  = (_Float16)a1[e];
    r[8 + e]  = (_Float16)b0[e];
    r[12 + e] = (_Float16)b1[e];
  }
  return r;
}

// epilogue: bias + sigmoid + 8-column mean -> g_block (gate never hits memory)
__device__ inline void sigmoid_reduce_store(v8f acc, const float* __restrict__ bg,
                                            float* __restrict__ gblk,
                                            int m_base, int n_base, int lane) {
  const int rsel = lane & 15, hi = lane >> 4;
  const float bval = bg[n_base + rsel];
#pragma unroll
  for (int v = 0; v < 8; ++v) {
    float z  = (acc[v] + bval) * GATE_SCALE;
    float gv = __builtin_amdgcn_rcpf(1.0f + __expf(-z));  // fast sigmoid
    gv += __shfl_xor(gv, 1, 32);
    gv += __shfl_xor(gv, 2, 32);
    gv += __shfl_xor(gv, 4, 32);
    if ((lane & 7) == 0) {
      int m  = m_base + v + 8 * hi;                 // C layout: M = v + 8*(L/16)
      int ng = (n_base >> 3) + (rsel >> 3);
      gblk[m * NB + ng] = gv * 0.125f;
    }
  }
}

// ---------------- kernel 1: gate GEMM via v_wmma_f32_16x16x32_f16 ----------
// 2x2 register-blocked: one wave owns a 32x32 gate tile (4 accumulators);
// each A/B fragment is reused twice -> halves L2 traffic vs 16x16/wave.
template <bool STAGED>
__global__ void gate_gemm_kernel(const _Float16* __restrict__ xh, const _Float16* __restrict__ Wh,
                                 const float* __restrict__ xr, const float* __restrict__ xi,
                                 const float* __restrict__ Wg, const float* __restrict__ bg,
                                 float* __restrict__ gblk) {
  const int lane = threadIdx.x & 31;
  const int wid  = blockIdx.x * (blockDim.x >> 5) + (threadIdx.x >> 5);
  const int nT2 = wid & 31, mT2 = wid >> 5;   // 32 N-supertiles x 64 M-supertiles
  const int m0 = mT2 * 32, n0 = nT2 * 32;
  const int rsel = lane & 15;
  const int hi   = lane >> 4;
  const int kA = hi * 8;                      // A frag: K runs at kA and kA+16
  const int kB = hi * 8;                      // B frag: K runs at kB and kB+16

  v8f acc00 = {}, acc01 = {}, acc10 = {}, acc11 = {};

#pragma unroll 2
  for (int k = 0; k < K2; k += 32) {
    v16h A0, A1, B0, B1;
    if (STAGED) {
      const _Float16* ar0 = xh + (size_t)(m0 + rsel)      * K2 + k;
      const _Float16* ar1 = xh + (size_t)(m0 + 16 + rsel) * K2 + k;
      const _Float16* br0 = Wh + (size_t)(n0 + rsel)      * K2 + k;
      const _Float16* br1 = Wh + (size_t)(n0 + 16 + rsel) * K2 + k;
      A0 = load_frag_f16(ar0 + kA, ar0 + kA + 16);
      A1 = load_frag_f16(ar1 + kA, ar1 + kA + 16);
      B0 = load_frag_f16(br0 + kB, br0 + kB + 16);
      B1 = load_frag_f16(br1 + kB, br1 + kB + 16);
    } else {
      const int ka0 = k + kA, ka1 = k + kA + 16;  // 8-aligned, never straddle D
      const int r0 = m0 + rsel, r1 = m0 + 16 + rsel;
      const float* pa00 = (ka0 < DD) ? (xr + (size_t)r0 * DD + ka0)
                                     : (xi + (size_t)r0 * DD + (ka0 - DD));
      const float* pa01 = (ka1 < DD) ? (xr + (size_t)r0 * DD + ka1)
                                     : (xi + (size_t)r0 * DD + (ka1 - DD));
      const float* pa10 = (ka0 < DD) ? (xr + (size_t)r1 * DD + ka0)
                                     : (xi + (size_t)r1 * DD + (ka0 - DD));
      const float* pa11 = (ka1 < DD) ? (xr + (size_t)r1 * DD + ka1)
                                     : (xi + (size_t)r1 * DD + (ka1 - DD));
      A0 = load_frag_f32(pa00, pa01);
      A1 = load_frag_f32(pa10, pa11);
      const float* pb0 = Wg + (size_t)(n0 + rsel)      * K2 + k + kB;
      const float* pb1 = Wg + (size_t)(n0 + 16 + rsel) * K2 + k + kB;
      B0 = load_frag_f32(pb0, pb0 + 16);
      B1 = load_frag_f32(pb1, pb1 + 16);
    }
    acc00 = __builtin_amdgcn_wmma_f32_16x16x32_f16(false, A0, false, B0, (short)0, acc00, false, false);
    acc01 = __builtin_amdgcn_wmma_f32_16x16x32_f16(false, A0, false, B1, (short)0, acc01, false, false);
    acc10 = __builtin_amdgcn_wmma_f32_16x16x32_f16(false, A1, false, B0, (short)0, acc10, false, false);
    acc11 = __builtin_amdgcn_wmma_f32_16x16x32_f16(false, A1, false, B1, (short)0, acc11, false, false);
  }

  sigmoid_reduce_store(acc00, bg, gblk, m0,      n0,      lane);
  sigmoid_reduce_store(acc01, bg, gblk, m0,      n0 + 16, lane);
  sigmoid_reduce_store(acc10, bg, gblk, m0 + 16, n0,      lane);
  sigmoid_reduce_store(acc11, bg, gblk, m0 + 16, n0 + 16, lane);
}

// ---------------- kernel 2: per-(b,s,n) Cayley transform + apply -----------
// One THREAD per (b,s,n). U x = (I+X)^{-1}(I-X) x = x + 2*sum_{k>=1} (-X)^k x
// with X = g*(A_r + i A_i), ||X|| <~ 0.12 -> 6 terms are exact to f32 eps.
// A's skew/symmetric parts staged once per block in LDS; every lane of a wave
// reads the same LDS address per step -> conflict-free broadcast.
__global__ void cayley_apply_kernel(const float* __restrict__ Areal,
                                    const float* __restrict__ Aimag,
                                    const float* __restrict__ gblk,
                                    const float* __restrict__ xr,
                                    const float* __restrict__ xi,
                                    float* __restrict__ out) {
  __shared__ float sAr[64];
  __shared__ float sAi[64];
  const int n = blockIdx.x;         // block index n (0..127)
  const int t = threadIdx.x;        // 256 threads

  if (t < 64) {                     // skew part of A_real
    int i = t >> 3, j = t & 7;
    sAr[t] = 0.5f * (Areal[n * 64 + i * 8 + j] - Areal[n * 64 + j * 8 + i]);
  } else if (t < 128) {             // symmetric part of A_imag
    int u = t - 64; int i = u >> 3, j = u & 7;
    sAi[u] = 0.5f * (Aimag[n * 64 + i * 8 + j] + Aimag[n * 64 + j * 8 + i]);
  }
  __syncthreads();

  const int m = blockIdx.y * blockDim.x + t;   // flattened (b,s) in [0,2048)
  const float g = gblk[m * NB + n];

  const size_t xoff = (size_t)m * DD + n * 8;
  v4f a = *(const v4f*)(xr + xoff);
  v4f b = *(const v4f*)(xr + xoff + 4);
  v4f c = *(const v4f*)(xi + xoff);
  v4f d = *(const v4f*)(xi + xoff + 4);

  float vr[8], vi[8], ur[8], ui[8];
#pragma unroll
  for (int i = 0; i < 4; ++i) {
    vr[i] = a[i]; vr[4 + i] = b[i];
    vi[i] = c[i]; vi[4 + i] = d[i];
  }
#pragma unroll
  for (int i = 0; i < 8; ++i) { ur[i] = vr[i]; ui[i] = vi[i]; }

  float s = -2.0f;
  for (int kt = 0; kt < NEUMANN_TERMS; ++kt) {
    float wr[8], wi[8];
#pragma unroll
    for (int i = 0; i < 8; ++i) {
      float accr = 0.0f, acci = 0.0f;
#pragma unroll
      for (int j = 0; j < 8; ++j) {
        const float ar = sAr[i * 8 + j];
        const float ai = sAi[i * 8 + j];
        accr = fmaf(ar, vr[j], accr);
        accr = fmaf(-ai, vi[j], accr);
        acci = fmaf(ar, vi[j], acci);
        acci = fmaf(ai, vr[j], acci);
      }
      wr[i] = g * accr;               // w = X v  (fold scalar g here)
      wi[i] = g * acci;
    }
#pragma unroll
    for (int i = 0; i < 8; ++i) {
      ur[i] = fmaf(s, wr[i], ur[i]);  // u += +-2 * X^k x
      ui[i] = fmaf(s, wi[i], ui[i]);
      vr[i] = wr[i]; vi[i] = wi[i];
    }
    s = -s;
  }

  float* orp = out + xoff;                    // real plane
  float* oip = out + (size_t)BSD + xoff;      // imag plane
  v4f o0, o1, o2, o3;
#pragma unroll
  for (int i = 0; i < 4; ++i) {
    o0[i] = ur[i]; o1[i] = ur[4 + i];
    o2[i] = ui[i]; o3[i] = ui[4 + i];
  }
  *(v4f*)orp       = o0;
  *(v4f*)(orp + 4) = o1;
  *(v4f*)oip       = o2;
  *(v4f*)(oip + 4) = o3;
}

// ---------------- launch ----------------
extern "C" void kernel_launch(void* const* d_in, const int* in_sizes, int n_in,
                              void* d_out, int out_size, void* d_ws, size_t ws_size,
                              hipStream_t stream) {
  const float* xr = (const float*)d_in[0];
  const float* xi = (const float*)d_in[1];
  const float* Ar = (const float*)d_in[2];
  const float* Ai = (const float*)d_in[3];
  const float* Wg = (const float*)d_in[4];
  const float* bg = (const float*)d_in[5];
  float* out = (float*)d_out;

  const size_t xh_elems = (size_t)BS * K2;   // 4Mi halves = 8 MiB
  const size_t wh_elems = (size_t)DD * K2;   // 2Mi halves = 4 MiB
  const size_t gb_bytes = (size_t)BS * NB * sizeof(float);  // 1 MiB
  const size_t staged_bytes = (xh_elems + wh_elems) * sizeof(_Float16) + gb_bytes;
  const bool staged = (ws_size >= staged_bytes);

  _Float16* xh = (_Float16*)d_ws;
  _Float16* Wh = xh + xh_elems;
  float* gblk = staged ? (float*)(Wh + wh_elems) : (float*)d_ws;

  // 2048 waves = 64 x 32 supertiles of 32x32; 8 waves per block
  if (staged) {
    cvt_x_kernel<<<(BS * K2) / 256, 256, 0, stream>>>(xr, xi, xh);
    cvt_w_kernel<<<(DD * K2) / 256, 256, 0, stream>>>(Wg, Wh);
    gate_gemm_kernel<true><<<256, 256, 0, stream>>>(xh, Wh, xr, xi, Wg, bg, gblk);
  } else {
    gate_gemm_kernel<false><<<256, 256, 0, stream>>>(xh, Wh, xr, xi, Wg, bg, gblk);
  }

  dim3 g2(NB, BS / 256);  // (128, 8) blocks x 256 threads -> one thread per (b,s,n)
  cayley_apply_kernel<<<g2, 256, 0, stream>>>(Ar, Ai, gblk, xr, xi, out);
}
